// OuterProductMean_8048768712948
// MI455X (gfx1250) — matrix operand order repaired and checked by reading, not compiled, not run on hardware
//
#include <hip/hip_runtime.h>
#include <hip/hip_bf16.h>

#define S_DIM 128
#define I_DIM 384
#define CM    256
#define CP    32
#define CZ    128
#define JBLK  16

typedef __attribute__((ext_vector_type(16))) __bf16 v16bf;
typedef __attribute__((ext_vector_type(8)))  float  v8f;
typedef __attribute__((ext_vector_type(4)))  unsigned int v4u;

union BF16Frag {
    v16bf v;
    v4u   q[2];
};

// ---- CDNA5 async memory->LDS helpers ---------------------------------------
// Per-lane 16B transfer, tracked by ASYNCcnt (no VGPR data round trip).
__device__ __forceinline__ void async_copy16(unsigned lds_off, const void* gaddr) {
    asm volatile("global_load_async_to_lds_b128 %0, %1, off"
                 :: "v"(lds_off), "v"(gaddr)
                 : "memory");
}

template <int N>
__device__ __forceinline__ void wait_async() {
#if __has_builtin(__builtin_amdgcn_s_wait_asynccnt)
    __builtin_amdgcn_s_wait_asynccnt(N);
#else
    asm volatile("s_wait_asynccnt %0" :: "n"(N) : "memory");
#endif
}

// ---------------------------------------------------------------------------
// Kernel 0: repack Wproj [128 z][1024 k] f32 -> bf16 fragment-ordered buffer
// wpack half index t = ((kt*8 + zt)*32 + lane)*16 + jj
//   lane: n' = lane&15 (z within tile), g = lane>>4
//   half jj holds B[k = kt*32 + g*16 + jj, z = zt*16 + n'] = Wproj[z][k]
// ---------------------------------------------------------------------------
__global__ __launch_bounds__(256) void pack_wproj(const float* __restrict__ Wproj,
                                                  __bf16* __restrict__ wpack) {
    int t    = blockIdx.x * 256 + threadIdx.x;   // 0 .. 131071
    int jj   = t & 15;
    int lane = (t >> 4) & 31;
    int zt   = (t >> 9) & 7;
    int kt   = t >> 12;
    int z = zt * 16 + (lane & 15);
    int k = kt * 32 + (lane >> 4) * 16 + jj;
    wpack[t] = (__bf16)Wproj[(size_t)z * (CP * CP) + k];
}

// ---------------------------------------------------------------------------
// Kernel 1: LayerNorm + dual projection. One wave32 per (s,i) row.
// Writes a_t/b_t transposed: [i][c (32)][s (128)] bf16 (K-major for WMMA).
// ---------------------------------------------------------------------------
__global__ __launch_bounds__(256) void ln_proj(const float* __restrict__ x,
                                               const float* __restrict__ gamma,
                                               const float* __restrict__ beta,
                                               const float* __restrict__ Wa,
                                               const float* __restrict__ Wb,
                                               __bf16* __restrict__ a_t,
                                               __bf16* __restrict__ b_t) {
    __shared__ float ysh[8][CM];
    const int tid  = threadIdx.x;
    const int lane = tid & 31;
    const int w    = tid >> 5;
    const int row  = blockIdx.x * 8 + w;          // 0 .. S*I-1, row = s*I + i
    const int s    = row / I_DIM;
    const int i    = row - s * I_DIM;

    const float* xr = x + (size_t)row * CM;
    float4 x0 = ((const float4*)xr)[lane * 2 + 0];
    float4 x1 = ((const float4*)xr)[lane * 2 + 1];

    float lsum = x0.x + x0.y + x0.z + x0.w + x1.x + x1.y + x1.z + x1.w;
    float lsq  = x0.x*x0.x + x0.y*x0.y + x0.z*x0.z + x0.w*x0.w
               + x1.x*x1.x + x1.y*x1.y + x1.z*x1.z + x1.w*x1.w;
    for (int off = 16; off > 0; off >>= 1) {
        lsum += __shfl_xor(lsum, off);
        lsq  += __shfl_xor(lsq,  off);
    }
    const float mu   = lsum * (1.0f / CM);
    const float var  = lsq * (1.0f / CM) - mu * mu;
    const float rstd = rsqrtf(var + 1e-5f);

    float4 g0 = ((const float4*)gamma)[lane * 2 + 0];
    float4 g1 = ((const float4*)gamma)[lane * 2 + 1];
    float4 e0 = ((const float4*)beta)[lane * 2 + 0];
    float4 e1 = ((const float4*)beta)[lane * 2 + 1];

    float4 y0, y1;
    y0.x = (x0.x - mu) * rstd * g0.x + e0.x;  y0.y = (x0.y - mu) * rstd * g0.y + e0.y;
    y0.z = (x0.z - mu) * rstd * g0.z + e0.z;  y0.w = (x0.w - mu) * rstd * g0.w + e0.w;
    y1.x = (x1.x - mu) * rstd * g1.x + e1.x;  y1.y = (x1.y - mu) * rstd * g1.y + e1.y;
    y1.z = (x1.z - mu) * rstd * g1.z + e1.z;  y1.w = (x1.w - mu) * rstd * g1.w + e1.w;
    ((float4*)ysh[w])[lane * 2 + 0] = y0;
    ((float4*)ysh[w])[lane * 2 + 1] = y1;
    __syncthreads();

    // lane p computes output channel p of both projections
    const float4* wa4 = (const float4*)(Wa + (size_t)lane * CM);
    const float4* wb4 = (const float4*)(Wb + (size_t)lane * CM);
    const float4* y4  = (const float4*)ysh[w];
    float aacc = 0.f, bacc = 0.f;
#pragma unroll 8
    for (int q = 0; q < CM / 4; ++q) {
        float4 yv = y4[q], av = wa4[q], bv = wb4[q];
        aacc += yv.x * av.x + yv.y * av.y + yv.z * av.z + yv.w * av.w;
        bacc += yv.x * bv.x + yv.y * bv.y + yv.z * bv.z + yv.w * bv.w;
    }
    const size_t o = ((size_t)i * CP + lane) * S_DIM + s;
    a_t[o] = (__bf16)aacc;
    b_t[o] = (__bf16)bacc;
}

// ---------------------------------------------------------------------------
// Kernel 2: fused outer-product-mean + final projection.
// Workgroup = (i, 16 consecutive j), 8 waves.
// Stage 2: P[c, (jl,d)] = sum_s a_t[i][c][s] * b[s, j0+jl, d]   (M=32,N=512,K=128)
//   b chunks staged via double-buffered async copies (ASYNCcnt), overlapping
//   the next chunk's HBM/L2 latency with this chunk's 8 WMMAs per wave.
// Stage 3: out[j, z]    = sum_cd outer_bf16[j][cd] * Wproj[z][cd] (M=16,N=128,K=1024)
// ---------------------------------------------------------------------------
__global__ __launch_bounds__(256) void fused_outer(const __bf16* __restrict__ a_t,
                                                   const __bf16* __restrict__ b_t,
                                                   const __bf16* __restrict__ wpack,
                                                   const float* __restrict__ bias,
                                                   float* __restrict__ out) {
    // Two 32 KB staging buffers [512 n][32 k] bf16 (64B rows).
    // Buffer 0 is reused after the K-loop as outer_bf16 [16 j][1024 cd] (2 KB rows).
    __shared__ __align__(32) unsigned char smem[2][JBLK * CP * CP * 2];

    const int jblk = blockIdx.x;
    const int i    = blockIdx.y;
    const int j0   = jblk * JBLK;
    const int tid  = threadIdx.x;
    const int lane = tid & 31;
    const int w    = tid >> 5;          // wave 0..7
    const int nloc = lane & 15;
    const int g    = lane >> 4;

    // ---- Stage 2: wave w owns n = [w*64, w*64+64), both m-tiles ----
    v8f acc[2][4];
#pragma unroll
    for (int mt = 0; mt < 2; ++mt)
#pragma unroll
        for (int nt = 0; nt < 4; ++nt) acc[mt][nt] = (v8f)0.f;

    // issue one chunk's worth of async copies (8 x b128 per thread)
    auto issue_chunk = [&](int s0, unsigned char* buf) {
        const unsigned base = (unsigned)(uintptr_t)buf;
#pragma unroll
        for (int r = 0; r < 8; ++r) {
            const int idx = r * 256 + tid;            // 0..2047 (= n*4 + q)
            const int n = idx >> 2;
            const int q = idx & 3;
            const __bf16* gp =
                b_t + ((size_t)(j0 * CP + n)) * S_DIM + s0 + q * 8;
            async_copy16(base + idx * 16, gp);
        }
    };

    issue_chunk(0, smem[0]);

    for (int ck = 0; ck < S_DIM / 32; ++ck) {
        unsigned char* cur = smem[ck & 1];
        const int s0 = ck * 32;
        if (ck < 3) {
            issue_chunk(s0 + 32, smem[(ck + 1) & 1]);
            wait_async<8>();   // current chunk's 8 copies complete (in-order)
        } else {
            wait_async<0>();
        }
        __syncthreads();       // all waves' pieces of chunk ck visible

        // A fragments (depend only on i): lane holds row c = mt*16 + nloc
        BF16Frag afrag[2];
#pragma unroll
        for (int mt = 0; mt < 2; ++mt) {
            const __bf16* ap = a_t + ((size_t)i * CP + mt * 16 + nloc) * S_DIM + s0;
            afrag[mt].q[0] = *(const v4u*)(ap + g * 8);
            afrag[mt].q[1] = *(const v4u*)(ap + 16 + g * 8);
        }
#pragma unroll
        for (int nt = 0; nt < 4; ++nt) {
            BF16Frag bfrag;
            const int n = w * 64 + nt * 16 + nloc;
            const unsigned char* bp = cur + n * 64 + g * 16;
            bfrag.q[0] = *(const v4u*)(bp);
            bfrag.q[1] = *(const v4u*)(bp + 32);
#pragma unroll
            for (int mt = 0; mt < 2; ++mt)
                acc[mt][nt] = __builtin_amdgcn_wmma_f32_16x16x32_bf16(
                    false, afrag[mt].v, false, bfrag.v, (short)0, acc[mt][nt],
                    false, false);
        }
        __syncthreads();       // chunk ck fully consumed before its buffer is reused
    }

    // ---- write scaled outer (1/S) as bf16 into LDS: [jl][c*32+d], row 2 KB ----
    const float scale = 1.0f / (float)S_DIM;
#pragma unroll
    for (int mt = 0; mt < 2; ++mt)
#pragma unroll
        for (int nt = 0; nt < 4; ++nt) {
            const int n  = w * 64 + nt * 16 + nloc;
            const int jl = n >> 5;
            const int d  = n & 31;
            __bf16* orow = (__bf16*)smem[0] + jl * (CP * CP) + d;
#pragma unroll
            for (int r = 0; r < 8; ++r) {
                const int c = mt * 16 + g * 8 + r;
                orow[c * 32] = (__bf16)(acc[mt][nt][r] * scale);
            }
        }
    __syncthreads();

    // ---- Stage 3: wave w owns z-tile w; M=16(j), K=1024, 32 WMMAs ----
    v8f acc3 = (v8f)0.f;
    for (int kt = 0; kt < CP * CP / 32; ++kt) {
        BF16Frag a3, b3;
        const __bf16* arow = (const __bf16*)smem[0] + nloc * (CP * CP) + kt * 32;
        a3.q[0] = *(const v4u*)(arow + g * 16);
        a3.q[1] = *(const v4u*)(arow + g * 16 + 8);
        const __bf16* wp = wpack + (((size_t)kt * 8 + w) * 32 + lane) * 16;
        b3.q[0] = *(const v4u*)(wp);
        b3.q[1] = *(const v4u*)(wp + 8);
        acc3 = __builtin_amdgcn_wmma_f32_16x16x32_bf16(
            false, a3.v, false, b3.v, (short)0, acc3, false, false);
    }

    // ---- epilogue: add bias, store f32 ----
    const int z  = w * 16 + nloc;
    const float bz = bias[z];
#pragma unroll
    for (int r = 0; r < 8; ++r) {
        const int j = j0 + g * 8 + r;
        out[((size_t)i * I_DIM + j) * CZ + z] = acc3[r] + bz;
    }
}

// ---------------------------------------------------------------------------
extern "C" void kernel_launch(void* const* d_in, const int* in_sizes, int n_in,
                              void* d_out, int out_size, void* d_ws, size_t ws_size,
                              hipStream_t stream) {
    const float* msa   = (const float*)d_in[0];
    const float* gamma = (const float*)d_in[1];
    const float* beta  = (const float*)d_in[2];
    const float* Wa    = (const float*)d_in[3];
    const float* Wb    = (const float*)d_in[4];
    const float* Wproj = (const float*)d_in[5];
    const float* bproj = (const float*)d_in[6];
    float* out = (float*)d_out;

    // workspace carve-up (bf16): a_t 3 MB, b_t 3 MB, wpack 256 KB
    __bf16* a_t   = (__bf16*)d_ws;
    __bf16* b_t   = a_t + (size_t)I_DIM * CP * S_DIM;
    __bf16* wpack = b_t + (size_t)I_DIM * CP * S_DIM;

    pack_wproj<<<dim3((CZ * CP * CP) / 256), 256, 0, stream>>>(Wproj, wpack);
    ln_proj<<<dim3((S_DIM * I_DIM) / 8), 256, 0, stream>>>(msa, gamma, beta, Wa, Wb,
                                                           a_t, b_t);
    fused_outer<<<dim3(I_DIM / JBLK, I_DIM), 256, 0, stream>>>(a_t, b_t, wpack,
                                                               bproj, out);
}
